// ODEFunc_8426725835111
// MI455X (gfx1250) — compile-verified, matrix operand order checked by reading, not run on hardware
//
#include <hip/hip_runtime.h>

typedef __attribute__((ext_vector_type(16))) __bf16 v16bf;
typedef __attribute__((ext_vector_type(8)))  __bf16 v8bf;
typedef __attribute__((ext_vector_type(8)))  float  v8f;

#define N_ 64
#define C_ 256
#define T_ 64
#define V_ 25

__global__ __launch_bounds__(256) void cvt_bf16_kernel(const float* __restrict__ src,
                                                       __bf16* __restrict__ dst, int n) {
  int i = blockIdx.x * 256 + threadIdx.x;
  if (i < n) dst[i] = (__bf16)src[i];
}

// One residual block: out = relu(W @ (A-diffused res) + b + res), per (n,t) tile.
// Safe to call with resIn == resOut (each workgroup fully reads its tile before writing).
__global__ __launch_bounds__(256) void ode_block_kernel(
    const float* resIn, float* resOut,
    const float* __restrict__ A,       // [V][V] fp32
    const __bf16* __restrict__ Wb,     // [C][C] bf16 row-major
    const float* __restrict__ bias) {  // [C]
  __shared__ float  res_s[C_][26];     // fp32 residual tile (padded stride)
  __shared__ __bf16 resb_s[C_][32];    // bf16 res, K-padded to 32 (u=25..31 zero)
  __shared__ __bf16 y_s[32][C_];       // y transposed [v][c]; rows 25..31 become zero
  __shared__ __bf16 Ab_s[32][32];      // bf16 A, zero-padded; Ab_s[v][u] = A[v][u]
  __shared__ float  b_s[C_];

  const int tid = threadIdx.x;
  const int n   = blockIdx.x / T_;
  const int t   = blockIdx.x % T_;

  b_s[tid] = bias[tid];
  // Stage zero-padded bf16 A (B-operand of diffusion GEMM reads Ab_s[v][u]).
#pragma unroll
  for (int i = tid; i < 32 * 32; i += 256) {
    const int row = i >> 5, col = i & 31;
    const float av = (row < V_ && col < V_) ? A[row * V_ + col] : 0.0f;
    Ab_s[row][col] = (__bf16)av;
  }

  // Load this thread's channel row (c = tid): 25 contiguous floats.
  const size_t rowBase = ((size_t)(n * C_ + tid) * T_ + t) * V_;
  float rr[V_];
#pragma unroll
  for (int u = 0; u < V_; ++u) rr[u] = resIn[rowBase + u];
#pragma unroll
  for (int u = 0; u < V_; ++u) {
    res_s[tid][u]  = rr[u];
    resb_s[tid][u] = (__bf16)rr[u];
  }
#pragma unroll
  for (int u = V_; u < 32; ++u) resb_s[tid][u] = (__bf16)0.0f;

  __syncthreads();  // staging complete

  const int wave  = tid >> 5;    // 0..7 -> M-tile pair {2w, 2w+1}
  const int lane  = tid & 31;
  const int lhalf = lane >> 4;   // selects K-chunk half per ISA 16-bit operand layout
  const int l16   = lane & 15;
  const int kb0   = lhalf * 8;

  // ---- Stage 1: graph diffusion via WMMA: y[c][v] = sum_u res[c][u] * A[v][u]
  // M = c (16 tiles), N = v (2 tiles), K = u (single 32-step, zero padded).
  {
    v16bf bA[2];
#pragma unroll
    for (int q = 0; q < 2; ++q) {
      const int v = q * 16 + l16;
      const v8bf* p = reinterpret_cast<const v8bf*>(&Ab_s[v][kb0]);
      v8bf lo = p[0];
      v8bf hi = p[2];
#pragma unroll
      for (int i = 0; i < 8; ++i) { bA[q][i] = lo[i]; bA[q][i + 8] = hi[i]; }
    }
    v16bf aR[2];
#pragma unroll
    for (int m = 0; m < 2; ++m) {
      const int c = (2 * wave + m) * 16 + l16;
      const v8bf* p = reinterpret_cast<const v8bf*>(&resb_s[c][kb0]);
      v8bf lo = p[0];
      v8bf hi = p[2];
#pragma unroll
      for (int i = 0; i < 8; ++i) { aR[m][i] = lo[i]; aR[m][i + 8] = hi[i]; }
    }
#pragma unroll
    for (int m = 0; m < 2; ++m) {
#pragma unroll
      for (int q = 0; q < 2; ++q) {
        v8f zc;
#pragma unroll
        for (int i = 0; i < 8; ++i) zc[i] = 0.0f;
        v8f y = __builtin_amdgcn_wmma_f32_16x16x32_bf16(
            false, aR[m], false, bA[q], (short)0, zc, false, false);
        // D layout: lane holds rows c = mt*16 + kb0 + 0..7 (consecutive!) of column v.
        v8bf yb;
#pragma unroll
        for (int i = 0; i < 8; ++i) yb[i] = (__bf16)y[i];
        const int v = q * 16 + l16;
        *reinterpret_cast<v8bf*>(&y_s[v][(2 * wave + m) * 16 + kb0]) = yb;
      }
    }
  }

  __syncthreads();  // y_s fully written (rows v>=25 are zero since Ab_s rows are zero)

  // ---- Stage 2: channel GEMM z[o][v] = sum_c W[o][c] * y[c][v]
  v8f acc[2][2];
#pragma unroll
  for (int m = 0; m < 2; ++m)
#pragma unroll
    for (int q = 0; q < 2; ++q)
#pragma unroll
      for (int i = 0; i < 8; ++i) acc[m][q][i] = 0.0f;

#pragma unroll
  for (int k = 0; k < 8; ++k) {
    const int kbase = k * 32 + kb0;

    v16bf afrag[2];
#pragma unroll
    for (int m = 0; m < 2; ++m) {
      const int o = (2 * wave + m) * 16 + l16;
      const v8bf* p = reinterpret_cast<const v8bf*>(Wb + (size_t)o * C_ + kbase);
      v8bf lo = p[0];
      v8bf hi = p[2];
#pragma unroll
      for (int i = 0; i < 8; ++i) { afrag[m][i] = lo[i]; afrag[m][i + 8] = hi[i]; }
    }

    v16bf bfrag[2];
#pragma unroll
    for (int q = 0; q < 2; ++q) {
      const int vrow = q * 16 + l16;
      const v8bf* p = reinterpret_cast<const v8bf*>(&y_s[vrow][kbase]);
      v8bf lo = p[0];
      v8bf hi = p[2];
#pragma unroll
      for (int i = 0; i < 8; ++i) { bfrag[q][i] = lo[i]; bfrag[q][i + 8] = hi[i]; }
    }

#pragma unroll
    for (int m = 0; m < 2; ++m)
#pragma unroll
      for (int q = 0; q < 2; ++q)
        acc[m][q] = __builtin_amdgcn_wmma_f32_16x16x32_bf16(
            false, afrag[m], false, bfrag[q], (short)0, acc[m][q], false, false);
  }

  // Epilogue: relu(z + b + res). C/D layout: VGPR r -> M = r + 8*lhalf, N = l16.
#pragma unroll
  for (int m = 0; m < 2; ++m) {
    const int mt = 2 * wave + m;
#pragma unroll
    for (int q = 0; q < 2; ++q) {
      const int v = q * 16 + l16;
      if (v < V_) {
#pragma unroll
        for (int r = 0; r < 8; ++r) {
          const int o = mt * 16 + lhalf * 8 + r;
          float val = acc[m][q][r] + b_s[o] + res_s[o][v];
          resOut[((size_t)(n * C_ + o) * T_ + t) * V_ + v] = fmaxf(val, 0.0f);
        }
      }
    }
  }
}

extern "C" void kernel_launch(void* const* d_in, const int* in_sizes, int n_in,
                              void* d_out, int out_size, void* d_ws, size_t ws_size,
                              hipStream_t stream) {
  (void)in_sizes; (void)n_in; (void)out_size; (void)ws_size;
  // setup_inputs order: t, x, A, w1, b1, w2, b2, w3, b3, w4, b4
  const float* x = (const float*)d_in[1];
  const float* A = (const float*)d_in[2];
  const float* w[4] = {(const float*)d_in[3], (const float*)d_in[5],
                       (const float*)d_in[7], (const float*)d_in[9]};
  const float* b[4] = {(const float*)d_in[4], (const float*)d_in[6],
                       (const float*)d_in[8], (const float*)d_in[10]};
  float* out = (float*)d_out;

  const int wn = C_ * C_;       // 65536 elements per weight matrix
  __bf16* wb0 = (__bf16*)d_ws;  // 4 * 65536 * 2B = 512 KB scratch

  for (int i = 0; i < 4; ++i)
    cvt_bf16_kernel<<<(wn + 255) / 256, 256, 0, stream>>>(
        w[i], wb0 + (size_t)i * wn, wn);

  dim3 grid(N_ * T_);
  ode_block_kernel<<<grid, 256, 0, stream>>>(x,   out, A, wb0 + 0 * (size_t)wn, b[0]);
  ode_block_kernel<<<grid, 256, 0, stream>>>(out, out, A, wb0 + 1 * (size_t)wn, b[1]);
  ode_block_kernel<<<grid, 256, 0, stream>>>(out, out, A, wb0 + 2 * (size_t)wn, b[2]);
  ode_block_kernel<<<grid, 256, 0, stream>>>(out, out, A, wb0 + 3 * (size_t)wn, b[3]);
}
